// VisionBDHv2_11991548690702
// MI455X (gfx1250) — matrix operand order repaired and checked by reading, not compiled, not run on hardware
//
#include <hip/hip_runtime.h>

typedef __bf16 bf16_t;
typedef __attribute__((ext_vector_type(16))) __bf16 bf16x16;
typedef __attribute__((ext_vector_type(8)))  float  floatx8;
typedef __attribute__((ext_vector_type(4)))  unsigned int uint4v;

// ---------- bf16 <-> f32 helpers (bit-exact RNE, no reliance on __bf16 casts) ----------
__device__ __forceinline__ bf16_t f2bf(float f) {
    union { float f; unsigned u; } x; x.f = f;
    unsigned r = x.u + 0x7FFFu + ((x.u >> 16) & 1u);
    union { unsigned short s; bf16_t b; } y; y.s = (unsigned short)(r >> 16);
    return y.b;
}
__device__ __forceinline__ float bf2f(bf16_t b) {
    union { unsigned short s; bf16_t b; } y; y.b = b;
    union { unsigned u; float f; } x; x.u = ((unsigned)y.s) << 16;
    return x.f;
}
__device__ __forceinline__ floatx8 fzero8() {
    floatx8 v;
#pragma unroll
    for (int i = 0; i < 8; ++i) v[i] = 0.0f;
    return v;
}
__device__ __forceinline__ long boff(int z, int dv, long s1, long s2) {
    return (long)(z / dv) * s1 + (long)(z % dv) * s2;
}

enum { EP_F32 = 0, EP_RELU_BF16 = 1, EP_RELU_MUL = 2, EP_BIAS = 3, EP_EMBED = 4 };

// ---------------------------------------------------------------------------
// Batched bf16 GEMM, f32 accumulate via v_wmma_f32_16x16x32_bf16.
// C[m,n] = sum_k A[m,k] * B[n,k]   (B is k-contiguous, i.e. "transposed")
// Block tile 64(M) x 128(N), 8 waves, each wave a 32x32 tile (2x2 WMMA accs).
// REQUIREMENTS: K % 32 == 0; lda, ldb and all batch offsets multiples of 8;
// A readable for ceil(M/64)*64 rows, B readable for ceil(N/128)*128 rows
// (overread rows may be garbage: the epilogue only stores gm<M, gn<N; buffers
// carry slack bytes so the reads stay in-bounds).
// ---------------------------------------------------------------------------
template<int EP>
__global__ __launch_bounds__(256) void gemm_kernel(
    const bf16_t* __restrict__ A, long sA1, long sA2, int dA, int lda,
    const bf16_t* __restrict__ Bm, long sB1, long sB2, int dB, int ldb,
    void* __restrict__ Ov, long sO1, long sO2, int dO, int ldo,
    const bf16_t* __restrict__ aux,        // EP_RELU_MUL (same layout as O)
    const float* __restrict__ bias,        // EP_BIAS / EP_EMBED
    const float* __restrict__ pos, int TT, // EP_EMBED
    int M, int N, int K)
{
    __shared__ __align__(16) bf16_t As[64][32];
    __shared__ __align__(16) bf16_t Bs[128][32];   // Bs[n][k]

    const int z = blockIdx.z;
    A  += boff(z, dA, sA1, sA2);
    Bm += boff(z, dB, sB1, sB2);
    const long obase = boff(z, dO, sO1, sO2);

    const int tid  = threadIdx.x;
    const int lane = tid & 31;
    const int wave = tid >> 5;
    const int wm   = wave >> 2;       // 0..1
    const int wn   = wave & 3;        // 0..3
    const int m0   = blockIdx.x * 64;
    const int n0   = blockIdx.y * 128;
    const int r16  = lane & 15;
    const int hi   = lane >> 4;       // 0/1

    const int srow = tid >> 2;        // 0..63 (staging row)
    const int skg  = (tid & 3) << 3;  // 0,8,16,24 (staging k-group)

    floatx8 acc[2][2];
#pragma unroll
    for (int i = 0; i < 2; ++i)
#pragma unroll
        for (int j = 0; j < 2; ++j) acc[i][j] = fzero8();

    const bf16_t* Ap  = A  + (long)(m0 + srow) * lda + skg;
    const bf16_t* Bp0 = Bm + (long)(n0 + srow) * ldb + skg;
    const bf16_t* Bp1 = Bm + (long)(n0 + 64 + srow) * ldb + skg;

    for (int kk = 0; kk < K; kk += 32) {
        if (kk + 32 < K) {   // uniform scalar branch only
            __builtin_prefetch(Ap + kk + 32, 0, 3);
            __builtin_prefetch(Bp0 + kk + 32, 0, 3);
            __builtin_prefetch(Bp1 + kk + 32, 0, 3);
        }
        // stage A + B: three unguarded 16B loads + 16B LDS stores per thread
        uint4v va = *(const uint4v*)(Ap + kk);
        uint4v v0 = *(const uint4v*)(Bp0 + kk);
        uint4v v1 = *(const uint4v*)(Bp1 + kk);
        *(uint4v*)(&As[srow][skg])      = va;
        *(uint4v*)(&Bs[srow][skg])      = v0;
        *(uint4v*)(&Bs[64 + srow][skg]) = v1;
        __syncthreads();

        // A fragment (16-bit A 16x32): lanes 0-15 K=0..7 & 16..23, lanes 16-31 K=8..15 & 24..31
        bf16x16 af[2], bfv[2];
#pragma unroll
        for (int i = 0; i < 2; ++i) {
            int row = wm * 32 + i * 16 + r16;
            int kb = hi * 8;
#pragma unroll
            for (int e = 0; e < 8; ++e) {
                af[i][e]     = As[row][kb + e];
                af[i][e + 8] = As[row][kb + 16 + e];
            }
        }
        // B fragment (16-bit B 32x16): lane col = lane%16, lanes 0-15 K=0..7 & 16..23, lanes 16-31 K=8..15 & 24..31
#pragma unroll
        for (int j = 0; j < 2; ++j) {
            int col = wn * 32 + j * 16 + r16;
            int kb = hi * 8;
#pragma unroll
            for (int e = 0; e < 16; ++e) bfv[j][e] = Bs[col][kb + (e & 7) + ((e >> 3) << 4)];
        }
#pragma unroll
        for (int i = 0; i < 2; ++i)
#pragma unroll
            for (int j = 0; j < 2; ++j)
                acc[i][j] = __builtin_amdgcn_wmma_f32_16x16x32_bf16(
                    false, af[i], false, bfv[j], (short)0, acc[i][j], false, false);
        __syncthreads();
    }

    // epilogue: C/D layout: VGPR v -> row v + (lane/16)*8, col lane%16
#pragma unroll
    for (int i = 0; i < 2; ++i)
#pragma unroll
        for (int j = 0; j < 2; ++j)
#pragma unroll
            for (int v = 0; v < 8; ++v) {
                int gm = m0 + wm * 32 + i * 16 + hi * 8 + v;
                int gn = n0 + wn * 32 + j * 16 + r16;
                if (gm < M && gn < N) {
                    float a = acc[i][j][v];
                    long oidx = obase + (long)gm * ldo + gn;
                    if (EP == EP_F32) {
                        ((float*)Ov)[oidx] = a;
                    } else if (EP == EP_RELU_BF16) {
                        ((bf16_t*)Ov)[oidx] = f2bf(a > 0.0f ? a : 0.0f);
                    } else if (EP == EP_RELU_MUL) {
                        float xs = bf2f(aux[oidx]);
                        ((bf16_t*)Ov)[oidx] = f2bf((a > 0.0f ? a : 0.0f) * xs);
                    } else if (EP == EP_BIAS) {
                        ((float*)Ov)[oidx] = a + bias[gn];
                    } else { // EP_EMBED
                        ((float*)Ov)[oidx] = a + bias[gn] + pos[(long)(gm % TT) * ldo + gn];
                    }
                }
            }
}

// ---------------------------------------------------------------------------
// Small helper kernels
// ---------------------------------------------------------------------------
__global__ void cvt_kernel(const float* __restrict__ in, bf16_t* __restrict__ out, long n) {
    long i = (long)blockIdx.x * 256 + threadIdx.x;
    if (i < n) out[i] = f2bf(in[i]);
}

// out[b][c][r] = in[b][r][c]  (convert f32 -> bf16 with transpose of last two dims)
__global__ void cvtT_kernel(const float* __restrict__ in, bf16_t* __restrict__ out,
                            int R, int C, long total) {
    long i = (long)blockIdx.x * 256 + threadIdx.x;
    if (i >= total) return;
    int c = (int)(i % C);
    long rest = i / C;
    int r = (int)(rest % R);
    long b = rest / R;
    out[(b * C + c) * (long)R + r] = f2bf(in[i]);
}

__global__ void im2col_kernel(const float* __restrict__ x, bf16_t* __restrict__ out) {
    long i = (long)blockIdx.x * 256 + threadIdx.x;
    const long total = (long)12544 * 768;
    if (i >= total) return;
    int k = (int)(i % 768);
    int m = (int)(i / 768);
    int b = m / 196, t = m % 196;
    int hp = t / 14, wp = t % 14;
    int c = k >> 8, r = k & 255;
    int p = r >> 4, q = r & 15;
    long xi = (((long)(b * 3 + c) * 224) + hp * 16 + p) * 224 + wp * 16 + q;
    out[i] = f2bf(x[xi]);
}

// zero the token-pad region of hnT: hnT[b][d][196..223] = 0
__global__ void zero_hnT_kernel(bf16_t* __restrict__ hnT) {
    long i = (long)blockIdx.x * 256 + threadIdx.x;
    const long total = (long)64 * 768 * 28;
    if (i >= total) return;
    int t = 196 + (int)(i % 28);
    long rest = i / 28;          // b*768 + d
    hnT[rest * 224 + t] = f2bf(0.0f);
}

__device__ __forceinline__ float block_sum(float v, float* sm) {
    int t = threadIdx.x;
    sm[t] = v; __syncthreads();
    for (int s = 128; s > 0; s >>= 1) {
        if (t < s) sm[t] += sm[t + s];
        __syncthreads();
    }
    float r = sm[0]; __syncthreads();
    return r;
}

// LayerNorm over rows of 768. Optional outputs: f32, bf16, and transposed bf16
// (obT layout (B, 768, 224): obT[((b*768)+d)*224 + t], row = b*196 + t).
__global__ __launch_bounds__(256) void ln768_kernel(const float* __restrict__ in,
        float* __restrict__ of, bf16_t* __restrict__ ob, bf16_t* __restrict__ obT) {
    long row = blockIdx.x;
    int t = threadIdx.x;
    __shared__ float sm[256];
    const float* p = in + row * 768;
    float v[3]; float s = 0.0f;
#pragma unroll
    for (int i = 0; i < 3; ++i) { v[i] = p[t + i * 256]; s += v[i]; }
    float mean = block_sum(s, sm) * (1.0f / 768.0f);
    float sq = 0.0f;
#pragma unroll
    for (int i = 0; i < 3; ++i) { v[i] -= mean; sq += v[i] * v[i]; }
    float inv = rsqrtf(block_sum(sq, sm) * (1.0f / 768.0f) + 1e-5f);
    long b = row / 196, tt = row % 196;
#pragma unroll
    for (int i = 0; i < 3; ++i) {
        float r = v[i] * inv;
        long idx = row * 768 + t + i * 256;
        if (of) of[idx] = r;
        if (ob) ob[idx] = f2bf(r);
        if (obT) obT[(b * 768 + t + i * 256) * 224 + tt] = f2bf(r);
    }
}

// h = LN(hn + LN(yMLP)), rows of 768
__global__ __launch_bounds__(256) void ln_combine_kernel(const float* __restrict__ ymlp,
        const float* __restrict__ hn, float* __restrict__ hout) {
    long row = blockIdx.x;
    int t = threadIdx.x;
    __shared__ float sm[256];
    const float* yp = ymlp + row * 768;
    const float* hp = hn + row * 768;
    float y[3]; float s = 0.0f;
#pragma unroll
    for (int i = 0; i < 3; ++i) { y[i] = yp[t + i * 256]; s += y[i]; }
    float mean = block_sum(s, sm) * (1.0f / 768.0f);
    float sq = 0.0f;
#pragma unroll
    for (int i = 0; i < 3; ++i) { y[i] -= mean; sq += y[i] * y[i]; }
    float inv = rsqrtf(block_sum(sq, sm) * (1.0f / 768.0f) + 1e-5f);
    float w[3]; s = 0.0f;
#pragma unroll
    for (int i = 0; i < 3; ++i) { w[i] = hp[t + i * 256] + y[i] * inv; s += w[i]; }
    mean = block_sum(s, sm) * (1.0f / 768.0f);
    sq = 0.0f;
#pragma unroll
    for (int i = 0; i < 3; ++i) { w[i] -= mean; sq += w[i] * w[i]; }
    inv = rsqrtf(block_sum(sq, sm) * (1.0f / 768.0f) + 1e-5f);
#pragma unroll
    for (int i = 0; i < 3; ++i) hout[row * 768 + t + i * 256] = w[i] * inv;
}

// RoPE on x_sparse (layout (B,T,nh,N) with N=512), qr = x*cos + rot(x)*sin
__global__ void rope_kernel(const bf16_t* __restrict__ xs, const float* __restrict__ freqs,
                            bf16_t* __restrict__ qr, long total) {
    long i = (long)blockIdx.x * 256 + threadIdx.x;
    if (i >= total) return;
    int n = (int)(i % 512);
    int t = (int)((i / (512 * 12)) % 196);
    float x = bf2f(xs[i]);
    float xr = (n & 1) ? bf2f(xs[i - 1]) : -bf2f(xs[i + 1]);
    float ph = (float)t * freqs[n];
    ph -= floorf(ph);
    float a = ph * 6.28318530717958647692f;
    qr[i] = f2bf(x * __cosf(a) + xr * __sinf(a));
}

// softmax over rows of 196 (scale 1/sqrt(512)); f32 in (ld 196) -> bf16 out (ld 224,
// pad cols 196..223 written as zero so the next GEMM can use K=224 unguarded)
__global__ __launch_bounds__(256) void softmax_kernel(const float* __restrict__ in,
                                                      bf16_t* __restrict__ out) {
    long row = blockIdx.x;
    int t = threadIdx.x;
    const float scale = 0.04419417382415922f; // 1/sqrt(512)
    const float* p = in + row * 196;
    __shared__ float sm[256];
    float v = (t < 196) ? p[t] * scale : -3.4e38f;
    sm[t] = v; __syncthreads();
    for (int s = 128; s > 0; s >>= 1) {
        if (t < s) sm[t] = fmaxf(sm[t], sm[t + s]);
        __syncthreads();
    }
    float mx = sm[0]; __syncthreads();
    float e = (t < 196) ? __expf(v - mx) : 0.0f;
    float tot = block_sum(e, sm);
    if (t < 224) out[row * 224 + t] = f2bf((t < 196) ? e / tot : 0.0f);
}

// mean over T then LN, per batch image -> pooled bf16 (64 x 768)
__global__ __launch_bounds__(256) void pool_kernel(const float* __restrict__ hsrc,
                                                   bf16_t* __restrict__ outb) {
    int b = blockIdx.x;
    int t = threadIdx.x;
    __shared__ float sm[256];
    const float* bp = hsrc + (long)b * 196 * 768;
    float v[3] = {0.0f, 0.0f, 0.0f};
    for (int s = 0; s < 196; ++s) {
        const float* rp = bp + (long)s * 768;
        v[0] += rp[t]; v[1] += rp[t + 256]; v[2] += rp[t + 512];
    }
    float ssum = 0.0f;
#pragma unroll
    for (int i = 0; i < 3; ++i) { v[i] *= (1.0f / 196.0f); ssum += v[i]; }
    float mean = block_sum(ssum, sm) * (1.0f / 768.0f);
    float sq = 0.0f;
#pragma unroll
    for (int i = 0; i < 3; ++i) { v[i] -= mean; sq += v[i] * v[i]; }
    float inv = rsqrtf(block_sum(sq, sm) * (1.0f / 768.0f) + 1e-5f);
#pragma unroll
    for (int i = 0; i < 3; ++i) outb[(long)b * 768 + t + i * 256] = f2bf(v[i] * inv);
}

// ---------------------------------------------------------------------------
// Host-side launch helper
// ---------------------------------------------------------------------------
template<int EP>
static void launch_gemm(hipStream_t s,
    const bf16_t* A, long sA1, long sA2, int dA, int lda,
    const bf16_t* B, long sB1, long sB2, int dB, int ldb,
    void* O, long sO1, long sO2, int dO, int ldo,
    const bf16_t* aux, const float* bias, const float* pos, int TT,
    int M, int N, int K, int nb)
{
    dim3 g((M + 63) / 64, (N + 127) / 128, nb);
    gemm_kernel<EP><<<g, dim3(256), 0, s>>>(A, sA1, sA2, dA, lda,
                                            B, sB1, sB2, dB, ldb,
                                            O, sO1, sO2, dO, ldo,
                                            aux, bias, pos, TT, M, N, K);
}

extern "C" void kernel_launch(void* const* d_in, const int* in_sizes, int n_in,
                              void* d_out, int out_size, void* d_ws, size_t ws_size,
                              hipStream_t stream) {
    (void)in_sizes; (void)n_in; (void)out_size; (void)ws_size;
    const float* x        = (const float*)d_in[0];
    const float* conv_w   = (const float*)d_in[1];
    const float* conv_b   = (const float*)d_in[2];
    const float* pos_emb  = (const float*)d_in[3];
    const float* encoder  = (const float*)d_in[4];
    const float* encoder_v= (const float*)d_in[5];
    const float* decoder  = (const float*)d_in[6];
    const float* freqs    = (const float*)d_in[7];
    const float* head_w   = (const float*)d_in[8];
    const float* head_b   = (const float*)d_in[9];

    const int B = 64, T = 196, Tp = 224, D = 768, NH = 12, NN = 512, NCLS = 1000, K1 = 768;
    const long BT = (long)B * T;          // 12544 (multiple of 64)

    char* base = (char*)d_ws;
    size_t off = 0;
    auto alloc = [&](size_t bytes) -> void* {
        off = (off + 255) & ~(size_t)255;
        void* p = base + off;
        off += bytes;
        return p;
    };
    float*  h_f   = (float*) alloc(BT * D * 4);
    float*  hn_f  = (float*) alloc(BT * D * 4);
    bf16_t* hn_b  = (bf16_t*)alloc(BT * D * 2);
    bf16_t* hnT   = (bf16_t*)alloc((long)B * D * Tp * 2);          // (B, 768, 224), pads zeroed
    bf16_t* a_im  = (bf16_t*)alloc(BT * (long)K1 * 2);
    bf16_t* xsp   = (bf16_t*)alloc(BT * NH * (long)NN * 2);
    bf16_t* qr    = (bf16_t*)alloc(BT * NH * (long)NN * 2 + (2 << 20)); // +2MB tile-overread slack
    float*  sc_f  = (float*) alloc((long)B * NH * T * T * 4);
    float*  ykv_f = (float*) alloc((long)B * NH * T * D * 4);
    bf16_t* ykvn  = (bf16_t*)alloc((long)B * NH * T * D * 2 + (1 << 17)); // +128KB slack
    float*  ymlp  = (float*) alloc(BT * D * 4);
    bf16_t* encT  = (bf16_t*)alloc((long)NH * D * NN * 2);         // (nh, 512, 768)
    bf16_t* encvT = (bf16_t*)alloc((long)NH * D * NN * 2);         // (nh, 512, 768)
    bf16_t* decT  = (bf16_t*)alloc((long)NH * NN * D * 2);         // (768, 6144)
    bf16_t* cw_b  = (bf16_t*)alloc((long)D * K1 * 2);              // (768, 768) already n-major
    bf16_t* hw_b  = (bf16_t*)alloc((long)NCLS * D * 2 + (1 << 16));// (1000, 768) +64KB slack
    bf16_t* pool_b= (bf16_t*)alloc((long)B * D * 2);
    bf16_t* sc_b  = qr;              // alias: qr dead after scores GEMM; ld 224
    bf16_t* zbuf  = (bf16_t*)ykv_f;  // alias: yKV f32 dead after its LN

    auto cvt = [&](const float* src, bf16_t* dst, long n) {
        cvt_kernel<<<(unsigned)((n + 255) / 256), 256, 0, stream>>>(src, dst, n);
    };
    auto cvtT = [&](const float* src, bf16_t* dst, int R, int C, long n) {
        cvtT_kernel<<<(unsigned)((n + 255) / 256), 256, 0, stream>>>(src, dst, R, C, n);
    };
    cvt (conv_w,    cw_b,  (long)D * K1);
    cvtT(encoder,   encT,  D, NN, (long)NH * D * NN);
    cvtT(encoder_v, encvT, D, NN, (long)NH * D * NN);
    cvtT(decoder,   decT,  NH * NN, D, (long)NH * NN * D);
    cvt (head_w,    hw_b,  (long)NCLS * D);
    zero_hnT_kernel<<<(unsigned)(((long)B * D * 28 + 255) / 256), 256, 0, stream>>>(hnT);

    // Patch embed: im2col + GEMM(A @ conv_w^T) + bias + pos_embed -> h
    im2col_kernel<<<(unsigned)((BT * K1 + 255) / 256), 256, 0, stream>>>(x, a_im);
    launch_gemm<EP_EMBED>(stream,
        a_im, 0, 0, 1, K1,
        cw_b, 0, 0, 1, K1,
        h_f,  0, 0, 1, D,
        nullptr, conv_b, pos_emb, T,
        (int)BT, D, K1, 1);

    for (int layer = 0; layer < 6; ++layer) {
        // hn = LN(h)  (f32 + bf16 + transposed-padded bf16)
        ln768_kernel<<<(unsigned)BT, 256, 0, stream>>>(h_f, hn_f, hn_b, hnT);

        // x_sparse = relu(hn @ encoder[h])  -> bf16, layout (B,T,nh,N)
        launch_gemm<EP_RELU_BF16>(stream,
            hn_b, 0, 0, 1, D,
            encT, (long)NN * D, 0, 1, D,
            xsp, (long)NN, 0, 1, NH * NN,
            nullptr, nullptr, nullptr, 0,
            (int)BT, NN, D, NH);

        // qr = rope(x_sparse)
        long rtot = BT * NH * NN;
        rope_kernel<<<(unsigned)((rtot + 255) / 256), 256, 0, stream>>>(xsp, freqs, qr, rtot);

        // scores_raw = qr @ qr^T per (b,h)
        launch_gemm<EP_F32>(stream,
            qr, (long)T * NH * NN, (long)NN, NH, NH * NN,
            qr, (long)T * NH * NN, (long)NN, NH, NH * NN,
            sc_f, (long)T * T, 0, 1, T,
            nullptr, nullptr, nullptr, 0,
            T, T, NN, B * NH);

        // softmax(scale * scores) -> bf16 (ld 224, zero pads; aliases qr)
        softmax_kernel<<<(unsigned)((long)B * NH * T), 256, 0, stream>>>(sc_f, sc_b);

        // yKV_raw = scores @ hn per (b,h): A=(196x224 padded), B=hnT (768x224), K=224
        launch_gemm<EP_F32>(stream,
            sc_b, (long)T * Tp, 0, 1, Tp,
            hnT, (long)D * Tp, 0, NH, Tp,
            ykv_f, (long)T * D, 0, 1, D,
            nullptr, nullptr, nullptr, 0,
            T, D, Tp, B * NH);

        // ykvn = LN(yKV) -> bf16
        ln768_kernel<<<(unsigned)((long)B * NH * T), 256, 0, stream>>>(ykv_f, nullptr, ykvn, nullptr);

        // z = relu(ykvn @ encoder_v[h]) * x_sparse -> bf16 (B,T,nh,N) (aliases yKV f32)
        launch_gemm<EP_RELU_MUL>(stream,
            ykvn, (long)T * D, 0, 1, D,
            encvT, 0, (long)NN * D, NH, D,
            zbuf, (long)T * NH * NN, (long)NN, NH, NH * NN,
            xsp, nullptr, nullptr, 0,
            T, NN, D, B * NH);

        // yMLP_raw = z @ decoder
        launch_gemm<EP_F32>(stream,
            zbuf, 0, 0, 1, NH * NN,
            decT, 0, 0, 1, NH * NN,
            ymlp, 0, 0, 1, D,
            nullptr, nullptr, nullptr, 0,
            (int)BT, D, NH * NN, 1);

        // h = LN(hn + LN(yMLP))
        ln_combine_kernel<<<(unsigned)BT, 256, 0, stream>>>(ymlp, hn_f, h_f);
    }

    // pooled = LN(mean_t h) -> bf16
    pool_kernel<<<B, 256, 0, stream>>>(h_f, pool_b);

    // out = pooled @ head_w^T + head_b  (f32)
    launch_gemm<EP_BIAS>(stream,
        pool_b, 0, 0, 1, D,
        hw_b, 0, 0, 1, D,
        (float*)d_out, 0, 0, 1, NCLS,
        nullptr, head_b, nullptr, 0,
        B, NCLS, D, 1);
}
